// JetMoeMoA_25546465477253
// MI455X (gfx1250) — matrix operand order repaired and checked by reading, not compiled, not run on hardware
//
#include <hip/hip_runtime.h>
#include <cstdint>

#define NE   8
#define TOPK 2
#define BM   128
#define BN   128
#define BK   32
#define LSTR 40                 // LDS row stride in bf16 elems (80B, padded vs 64B)
#define TSZ  (BM * LSTR)        // one LDS tile in bf16 elems

typedef __bf16 bf16;
typedef __attribute__((ext_vector_type(16))) __bf16 v16bf;
typedef __attribute__((ext_vector_type(8)))  __bf16 v8bf;
typedef __attribute__((ext_vector_type(4)))  __bf16 v4bf;
typedef __attribute__((ext_vector_type(8)))  float  v8f;
typedef __attribute__((ext_vector_type(4)))  int    v4i;

#if __has_builtin(__builtin_amdgcn_global_load_async_to_lds_b128)
#define HAVE_ASYNC 1
#else
#define HAVE_ASYNC 0
#endif

__device__ __forceinline__ v16bf cat16(v8bf lo, v8bf hi) {
  return __builtin_shufflevector(lo, hi, 0,1,2,3,4,5,6,7,8,9,10,11,12,13,14,15);
}

// 16-byte global -> LDS copy: async engine (ASYNCcnt) if available, else load+ds_store
__device__ __forceinline__ void cp16(bf16* dst_lds, const bf16* src_g) {
#if HAVE_ASYNC
  __builtin_amdgcn_global_load_async_to_lds_b128(
      (__attribute__((address_space(1))) v4i*)(uintptr_t)src_g,
      (__attribute__((address_space(3))) v4i*)(uint32_t)(uintptr_t)dst_lds,
      0, 0);
#else
  *(v8bf*)dst_lds = *(const v8bf*)src_g;
#endif
}

__device__ __forceinline__ void wait_async() {
#if HAVE_ASYNC
#if __has_builtin(__builtin_amdgcn_s_wait_asynccnt)
  __builtin_amdgcn_s_wait_asynccnt(0);
#else
  asm volatile("s_wait_asynccnt 0" ::: "memory");
#endif
#endif
}

// ---------------- init: zero counters + 256B zero-pad row ----------------
__global__ void moa_init(int* counts, int* cursor, int* zpad) {
  int i = threadIdx.x;
  if (i < NE) { counts[i] = 0; cursor[i] = 0; }
  if (i < 64) zpad[i] = 0;
}

// ---------------- f32 -> bf16 convert ----------------
__global__ void moa_cvt_bf16(const float* __restrict__ in, bf16* __restrict__ out, long long n) {
  long long i = ((long long)blockIdx.x * blockDim.x + threadIdx.x) * 4;
  if (i >= n) return;
  const float4 v = *(const float4*)(in + i);
  v4bf o;
  o[0] = (bf16)v.x; o[1] = (bf16)v.y; o[2] = (bf16)v.z; o[3] = (bf16)v.w;
  *(v4bf*)(out + i) = o;
}

// ---------------- router: one wave per token ----------------
__global__ void moa_router(const float* __restrict__ x, const float* __restrict__ wg,
                           float* __restrict__ logits, float* __restrict__ gateArr,
                           int* __restrict__ eArr, int* __restrict__ counts,
                           int T, int D) {
  int gw   = (int)((blockIdx.x * (long long)blockDim.x + threadIdx.x) >> 5);
  int lane = threadIdx.x & 31;
  if (gw >= T) return;
  const float* xr = x + (long long)gw * D;
  float acc[NE];
#pragma unroll
  for (int e = 0; e < NE; ++e) acc[e] = 0.f;
  for (int d = lane; d < D; d += 32) {
    float xv = xr[d];
#pragma unroll
    for (int e = 0; e < NE; ++e) acc[e] += xv * wg[e * D + d];
  }
#pragma unroll
  for (int e = 0; e < NE; ++e) {
    float v = acc[e];
#pragma unroll
    for (int s = 16; s; s >>= 1) v += __shfl_xor(v, s, 32);
    acc[e] = v;
  }
  if (lane == 0) {
#pragma unroll
    for (int e = 0; e < NE; ++e) logits[(long long)gw * NE + e] = acc[e];
    int i0 = 0; float l0 = acc[0];
#pragma unroll
    for (int e = 1; e < NE; ++e) if (acc[e] > l0) { l0 = acc[e]; i0 = e; }
    int i1 = -1; float l1 = -3.4e38f;
#pragma unroll
    for (int e = 0; e < NE; ++e) if (e != i0 && acc[e] > l1) { l1 = acc[e]; i1 = e; }
    float ex  = __expf(l1 - l0);           // <= 1, stable
    float inv = 1.f / (1.f + ex);
    gateArr[2 * gw]     = inv;
    gateArr[2 * gw + 1] = ex * inv;
    eArr[2 * gw]     = i0;
    eArr[2 * gw + 1] = i1;
    atomicAdd(&counts[i0], 1);
    atomicAdd(&counts[i1], 1);
  }
}

// ---------------- exclusive scan over 8 experts ----------------
__global__ void moa_scan(const int* __restrict__ counts, int* __restrict__ offsets) {
  if (threadIdx.x == 0) {
    int s = 0;
    for (int e = 0; e < NE; ++e) { offsets[e] = s; s += counts[e]; }
  }
}

// ---------------- scatter pairs into expert segments ----------------
__global__ void moa_scatter(const int* __restrict__ eArr, const int* __restrict__ offsets,
                            int* __restrict__ cursor, int* __restrict__ perm, int P) {
  int p = blockIdx.x * blockDim.x + threadIdx.x;
  if (p >= P) return;
  int e = eArr[p];
  int pos = offsets[e] + atomicAdd(&cursor[e], 1);
  perm[pos] = p;
}

// ---------------- grouped GEMM 1: y = gather(x) @ w_in[e]^T  (bf16 out) ----------------
__global__ __launch_bounds__(256) void moa_gemm1(
    const bf16* __restrict__ xb,   // [T, D]
    const bf16* __restrict__ wb,   // [E, H, D]   (N=h rows, K=d contiguous)
    bf16* __restrict__ yb,         // [P, H] in segment order
    const int* __restrict__ perm,
    const int* __restrict__ offsets,
    const int* __restrict__ counts,
    const bf16* __restrict__ zpad, // 256B of zeros
    int D, int H)
{
  const int e = blockIdx.z;
  const int cnt = counts[e];
  const int mbase = blockIdx.y * BM;
  if (mbase >= cnt) return;
  const int start = offsets[e];
  const int nbase = blockIdx.x * BN;

  __shared__ __align__(16) bf16 As[2 * TSZ];
  __shared__ __align__(16) bf16 Bs[2 * TSZ];

  const int tid = threadIdx.x;
  const bf16* wbe = wb + (size_t)e * H * D;

  // ---- per-thread loop-invariant staging sources (2 A chunks + 2 B chunks) ----
  const bf16* asrc[2]; int astep[2]; int aldo[2];
  const bf16* bsrc[2];              int bldo[2];
#pragma unroll
  for (int c = 0; c < 2; ++c) {
    int idx = tid + c * 256;
    int r = idx >> 2;
    int off = (idx & 3) * 8;
    aldo[c] = r * LSTR + off;
    bldo[c] = r * LSTR + off;
    int m = mbase + r;
    if (m < cnt) { int t = perm[start + m] >> 1;
                   asrc[c] = xb + (size_t)t * D + off; astep[c] = BK; }
    else         { asrc[c] = zpad;                     astep[c] = 0;  }
    bsrc[c] = wbe + (size_t)(nbase + r) * D + off;
  }

  const int wave = tid >> 5, lane = tid & 31;
  const int wm = wave & 1, wn = wave >> 1;
  const int lrow = lane & 15;
  const int hi = (lane >= 16);

  v8f acc[4][2];
#pragma unroll
  for (int i = 0; i < 4; ++i)
#pragma unroll
    for (int j = 0; j < 2; ++j)
#pragma unroll
      for (int k = 0; k < 8; ++k) acc[i][j][k] = 0.f;

  // ---- prologue: stage buffer 0 ----
#pragma unroll
  for (int c = 0; c < 2; ++c) {
    cp16(As + aldo[c], asrc[c]); asrc[c] += astep[c];
    cp16(Bs + bldo[c], bsrc[c]); bsrc[c] += BK;
  }

  const int niter = D / BK;
  for (int it = 0; it < niter; ++it) {
    const int cur = it & 1;
    const int cb = cur * TSZ;
    wait_async();
    __syncthreads();                       // buf[cur] filled; buf[1-cur] free
    if (it + 1 < niter) {
      const int nb = (cur ^ 1) * TSZ;
#pragma unroll
      for (int c = 0; c < 2; ++c) {
        cp16(As + nb + aldo[c], asrc[c]); asrc[c] += astep[c];
        cp16(Bs + nb + bldo[c], bsrc[c]); bsrc[c] += BK;
      }
    }

    v16bf a[4], b[2];
#pragma unroll
    for (int mf = 0; mf < 4; ++mf) {       // A frag: lanes<16 K{0-7,16-23}, lanes>=16 K{8-15,24-31}
      const bf16* base = As + cb + (wm * 64 + mf * 16 + lrow) * LSTR;
      v8bf lo = *(const v8bf*)(base + (hi ? 8 : 0));
      v8bf hh = *(const v8bf*)(base + (hi ? 24 : 16));
      a[mf] = cat16(lo, hh);
    }
#pragma unroll
    for (int nf = 0; nf < 2; ++nf) {       // B frag: lanes<16 K{0-7,16-23}, lanes>=16 K{8-15,24-31}
      const bf16* base = Bs + cb + (wn * 32 + nf * 16 + lrow) * LSTR;
      v8bf lo = *(const v8bf*)(base + (hi ? 8 : 0));
      v8bf hh = *(const v8bf*)(base + (hi ? 24 : 16));
      b[nf] = cat16(lo, hh);
    }
#pragma unroll
    for (int mf = 0; mf < 4; ++mf)
#pragma unroll
      for (int nf = 0; nf < 2; ++nf)
        acc[mf][nf] = __builtin_amdgcn_wmma_f32_16x16x32_bf16(
            false, a[mf], false, b[nf], (short)0, acc[mf][nf], false, false);
  }

  const int moff = hi ? 8 : 0;             // C/D layout: M=j(+8), N=lane&15
#pragma unroll
  for (int mf = 0; mf < 4; ++mf)
#pragma unroll
    for (int nf = 0; nf < 2; ++nf) {
      int col = nbase + wn * 32 + nf * 16 + lrow;
#pragma unroll
      for (int j = 0; j < 8; ++j) {
        int m = mbase + wm * 64 + mf * 16 + moff + j;
        if (m < cnt)
          yb[(size_t)(start + m) * H + col] = (bf16)acc[mf][nf][j];
      }
    }
}

// ---------------- grouped GEMM 2: z[p] = gate[p] * (y @ w_out[e]^T)  (f32 out) ----------------
__global__ __launch_bounds__(256) void moa_gemm2(
    const bf16* __restrict__ yb,   // [P, H] segment order
    const bf16* __restrict__ wb,   // [E, D, H]   (N=d rows, K=h contiguous)
    float* __restrict__ zbuf,      // [P, D] indexed by pair id
    const int* __restrict__ perm,
    const int* __restrict__ offsets,
    const int* __restrict__ counts,
    const float* __restrict__ gateArr,
    const bf16* __restrict__ zpad,
    int D, int H)
{
  const int e = blockIdx.z;
  const int cnt = counts[e];
  const int mbase = blockIdx.y * BM;
  if (mbase >= cnt) return;
  const int start = offsets[e];
  const int nbase = blockIdx.x * BN;

  __shared__ __align__(16) bf16 As[2 * TSZ];
  __shared__ __align__(16) bf16 Bs[2 * TSZ];
  __shared__ int   pid[BM];
  __shared__ float gv[BM];

  const int tid = threadIdx.x;
  if (tid < BM) {
    int m = mbase + tid;
    if (m < cnt) { int p = perm[start + m]; pid[tid] = p; gv[tid] = gateArr[p]; }
    else         { pid[tid] = -1;           gv[tid] = 0.f; }
  }
  const bf16* wbe = wb + (size_t)e * D * H;

  const bf16* asrc[2]; int astep[2]; int aldo[2];
  const bf16* bsrc[2];              int bldo[2];
#pragma unroll
  for (int c = 0; c < 2; ++c) {
    int idx = tid + c * 256;
    int r = idx >> 2;
    int off = (idx & 3) * 8;
    aldo[c] = r * LSTR + off;
    bldo[c] = r * LSTR + off;
    int m = mbase + r;
    if (m < cnt) { asrc[c] = yb + (size_t)(start + m) * H + off; astep[c] = BK; }
    else         { asrc[c] = zpad;                               astep[c] = 0;  }
    bsrc[c] = wbe + (size_t)(nbase + r) * H + off;
  }

  const int wave = tid >> 5, lane = tid & 31;
  const int wm = wave & 1, wn = wave >> 1;
  const int lrow = lane & 15;
  const int hi = (lane >= 16);

  v8f acc[4][2];
#pragma unroll
  for (int i = 0; i < 4; ++i)
#pragma unroll
    for (int j = 0; j < 2; ++j)
#pragma unroll
      for (int k = 0; k < 8; ++k) acc[i][j][k] = 0.f;

#pragma unroll
  for (int c = 0; c < 2; ++c) {
    cp16(As + aldo[c], asrc[c]); asrc[c] += astep[c];
    cp16(Bs + bldo[c], bsrc[c]); bsrc[c] += BK;
  }

  const int niter = H / BK;
  for (int it = 0; it < niter; ++it) {
    const int cur = it & 1;
    const int cb = cur * TSZ;
    wait_async();
    __syncthreads();
    if (it + 1 < niter) {
      const int nb = (cur ^ 1) * TSZ;
#pragma unroll
      for (int c = 0; c < 2; ++c) {
        cp16(As + nb + aldo[c], asrc[c]); asrc[c] += astep[c];
        cp16(Bs + nb + bldo[c], bsrc[c]); bsrc[c] += BK;
      }
    }

    v16bf a[4], b[2];
#pragma unroll
    for (int mf = 0; mf < 4; ++mf) {
      const bf16* base = As + cb + (wm * 64 + mf * 16 + lrow) * LSTR;
      v8bf lo = *(const v8bf*)(base + (hi ? 8 : 0));
      v8bf hh = *(const v8bf*)(base + (hi ? 24 : 16));
      a[mf] = cat16(lo, hh);
    }
#pragma unroll
    for (int nf = 0; nf < 2; ++nf) {
      const bf16* base = Bs + cb + (wn * 32 + nf * 16 + lrow) * LSTR;
      v8bf lo = *(const v8bf*)(base + (hi ? 8 : 0));
      v8bf hh = *(const v8bf*)(base + (hi ? 24 : 16));
      b[nf] = cat16(lo, hh);
    }
#pragma unroll
    for (int mf = 0; mf < 4; ++mf)
#pragma unroll
      for (int nf = 0; nf < 2; ++nf)
        acc[mf][nf] = __builtin_amdgcn_wmma_f32_16x16x32_bf16(
            false, a[mf], false, b[nf], (short)0, acc[mf][nf], false, false);
  }

  const int moff = hi ? 8 : 0;
#pragma unroll
  for (int mf = 0; mf < 4; ++mf)
#pragma unroll
    for (int nf = 0; nf < 2; ++nf) {
      int col = nbase + wn * 32 + nf * 16 + lrow;
#pragma unroll
      for (int j = 0; j < 8; ++j) {
        int ml = wm * 64 + mf * 16 + moff + j;     // local row 0..127
        int p = pid[ml];
        if (p >= 0)
          zbuf[(size_t)p * D + col] = gv[ml] * acc[mf][nf][j];
      }
    }
}

// ---------------- fixed-order combine: out = z[2t] + z[2t+1] + bias ----------------
__global__ void moa_combine(const float* __restrict__ z, const float* __restrict__ bias,
                            float* __restrict__ out, int T, int D) {
  long long i = ((long long)blockIdx.x * blockDim.x + threadIdx.x) * 4;
  if (i >= (long long)T * D) return;
  int t = (int)(i / D);
  int d = (int)(i - (long long)t * D);
  const float4 a = *(const float4*)(z + (long long)(2 * t) * D + d);
  const float4 b = *(const float4*)(z + (long long)(2 * t + 1) * D + d);
  const float4 c = *(const float4*)(bias + d);
  float4 o;
  o.x = a.x + b.x + c.x; o.y = a.y + b.y + c.y;
  o.z = a.z + b.z + c.z; o.w = a.w + b.w + c.w;
  *(float4*)(out + i) = o;
}

extern "C" void kernel_launch(void* const* d_in, const int* in_sizes, int n_in,
                              void* d_out, int out_size, void* d_ws, size_t ws_size,
                              hipStream_t stream) {
  (void)n_in; (void)out_size; (void)ws_size;
  const float* x     = (const float*)d_in[0];
  const float* wg    = (const float*)d_in[1];
  const float* w_in  = (const float*)d_in[2];
  const float* w_out = (const float*)d_in[3];
  const float* bias  = (const float*)d_in[4];

  const int D = in_sizes[1] / NE;
  const int T = in_sizes[0] / D;
  const int H = in_sizes[2] / (NE * D);
  const int P = T * TOPK;

  float* out    = (float*)d_out;            // [T, D]
  float* logits = out + (size_t)T * D;      // [T, NE]

  char* ws = (char*)d_ws;
  size_t off = 0;
  auto carve = [&](size_t bytes) -> void* {
    off = (off + 255) & ~(size_t)255;
    void* p = ws + off;
    off += bytes;
    return p;
  };
  int*   counts  = (int*)carve(NE * 4);
  int*   cursor  = (int*)carve(NE * 4);
  int*   offsets = (int*)carve(NE * 4);
  int*   zpad    = (int*)carve(256);
  int*   eArr    = (int*)carve((size_t)P * 4);
  float* gateArr = (float*)carve((size_t)P * 4);
  int*   perm    = (int*)carve((size_t)P * 4);
  bf16*  xb      = (bf16*)carve((size_t)T * D * 2);
  bf16*  wib     = (bf16*)carve((size_t)NE * H * D * 2);
  bf16*  wob     = (bf16*)carve((size_t)NE * D * H * 2);
  bf16*  yb      = (bf16*)carve((size_t)P * H * 2);
  float* zbuf    = (float*)carve((size_t)P * D * 4);

  moa_init<<<1, 64, 0, stream>>>(counts, cursor, zpad);
  { long long n = (long long)T * D;
    moa_cvt_bf16<<<(unsigned)((n / 4 + 255) / 256), 256, 0, stream>>>(x, xb, n); }
  { long long n = (long long)NE * H * D;
    moa_cvt_bf16<<<(unsigned)((n / 4 + 255) / 256), 256, 0, stream>>>(w_in, wib, n); }
  { long long n = (long long)NE * D * H;
    moa_cvt_bf16<<<(unsigned)((n / 4 + 255) / 256), 256, 0, stream>>>(w_out, wob, n); }
  moa_router<<<(T * 32 + 255) / 256, 256, 0, stream>>>(x, wg, logits, gateArr, eArr, counts, T, D);
  moa_scan<<<1, 1, 0, stream>>>(counts, offsets);
  moa_scatter<<<(P + 255) / 256, 256, 0, stream>>>(eArr, offsets, cursor, perm, P);
  dim3 g1(H / BN, (T + BM - 1) / BM, NE);   // per-expert worst case cnt<=T
  moa_gemm1<<<g1, 256, 0, stream>>>(xb, wib, yb, perm, offsets, counts, (const bf16*)zpad, D, H);
  dim3 g2(D / BN, (T + BM - 1) / BM, NE);
  moa_gemm2<<<g2, 256, 0, stream>>>(yb, wob, zbuf, perm, offsets, counts, gateArr, (const bf16*)zpad, D, H);
  { long long n = (long long)T * D;
    moa_combine<<<(unsigned)((n / 4 + 255) / 256), 256, 0, stream>>>(zbuf, bias, out, T, D); }
}
